// LabelClassifier_82386062672565
// MI455X (gfx1250) — compile-verified, matrix operand order checked by reading, not run on hardware
//
#include <hip/hip_runtime.h>
#include <hip/hip_bf16.h>

// Problem constants (match reference)
#define S_  512
#define B_  32
#define H_  1024
#define L_  4096
#define D_  1024
#define V_  32000
#define K1_ 2048      // 2H
#define K2_ 1024      // D
#define NVT_ 400      // partial tiles over V (80 cols each)

typedef __attribute__((ext_vector_type(16))) __bf16 v16bf;
typedef __attribute__((ext_vector_type(8)))  float  v8f;
typedef int v4i_ __attribute__((vector_size(16)));   // matches builtin param type

#if defined(__has_builtin)
#if __has_builtin(__builtin_amdgcn_global_load_async_to_lds_b128) && \
    __has_builtin(__builtin_amdgcn_s_wait_asynccnt)
#define ASYNC_LDS 1
#endif
#endif
#ifndef ASYNC_LDS
#define ASYNC_LDS 0
#endif

#define GAS(p) ((__attribute__((address_space(1))) v4i_*)(p))
#define LAS(p) ((__attribute__((address_space(3))) v4i_*)(p))

__device__ __forceinline__ __bf16 f2bf(float f) {
    unsigned u = __builtin_bit_cast(unsigned, f);
    unsigned r = (u + 0x7fffu + ((u >> 16) & 1u)) >> 16;
    unsigned short h = (unsigned short)r;
    return __builtin_bit_cast(__bf16, h);
}
__device__ __forceinline__ float bf2f(__bf16 x) {
    unsigned short h = __builtin_bit_cast(unsigned short, x);
    unsigned u = ((unsigned)h) << 16;
    return __builtin_bit_cast(float, u);
}

__global__ void zero_out_kernel(float* out) { out[0] = 0.0f; }

__global__ void cvt_f32_bf16_kernel(const float* __restrict__ src,
                                    __bf16* __restrict__ dst, long n) {
    long i = (long)blockIdx.x * blockDim.x + threadIdx.x;
    long stride = (long)gridDim.x * blockDim.x;
    for (; i < n; i += stride) dst[i] = f2bf(src[i]);
}

// Build span_emb_bf16[L, 2H] = concat(hidden[beg,bid,:], hidden[endm1,bid,:])
__global__ void gather_kernel(const float* __restrict__ hidden,
                              const int* __restrict__ bid,
                              const int* __restrict__ beg,
                              const int* __restrict__ endm1,
                              __bf16* __restrict__ span_emb) {
    int l = blockIdx.x;
    int b  = bid[l];
    int s0 = beg[l];
    int s1 = endm1[l];
    const float* pb = hidden + ((size_t)s0 * B_ + b) * H_;
    const float* pe = hidden + ((size_t)s1 * B_ + b) * H_;
    __bf16* dst = span_emb + (size_t)l * K1_;
    for (int i = threadIdx.x; i < H_; i += blockDim.x) {
        dst[i]      = f2bf(pb[i]);
        dst[H_ + i] = f2bf(pe[i]);
    }
}

// feat_bf16[L, D] = tanh(span_emb @ W1^T + b1)
// grid (L/16, D/128), block 256 (8 waves). A tile staged in LDS in 1024-K chunks.
__global__ void __launch_bounds__(256, 1)
gemm1_kernel(const __bf16* __restrict__ A,
             const __bf16* __restrict__ W1b,
             const float* __restrict__ b1,
             __bf16* __restrict__ feat) {
    __shared__ __bf16 As[16 * 1024];       // 32 KB
    const int mt   = blockIdx.x;
    const int wave = threadIdx.x >> 5;
    const int lane = threadIdx.x & 31;
    const int n0   = (blockIdx.y * 8 + wave) * 16;

    v8f c = {};
    const int row  = lane & 15;
    const int koff = (lane < 16) ? 0 : 16;

    #pragma unroll 1
    for (int kc = 0; kc < K1_; kc += 1024) {
        // A-tile fill: 16 rows x 1024 cols bf16 (1024 x 32B chunks)
        const __bf16* srcA = A + (size_t)mt * 16 * K1_ + kc;
#if ASYNC_LDS
        #pragma unroll 1
        for (int i = threadIdx.x; i < 16 * 64; i += 256) {
            int r = i >> 6, ch = i & 63;
            __builtin_amdgcn_global_load_async_to_lds_b128(
                GAS(srcA + (size_t)r * K1_ + ch * 16),
                LAS(As + r * 1024 + ch * 16), 0, 0);
        }
        __builtin_amdgcn_s_wait_asynccnt(0);
#else
        #pragma unroll 1
        for (int i = threadIdx.x; i < 16 * 64; i += 256) {
            int r = i >> 6, ch = i & 63;
            ((v16bf*)As)[r * 64 + ch] = *(const v16bf*)(srcA + (size_t)r * K1_ + ch * 16);
        }
#endif
        __syncthreads();
        const __bf16* bp = W1b + (size_t)(n0 + row) * K1_ + kc + koff;
        #pragma unroll 2
        for (int k = 0; k < 1024; k += 32) {
            v16bf a = *(const v16bf*)(As + row * 1024 + k + koff);
            v16bf bm = *(const v16bf*)(bp + k);
            c = __builtin_amdgcn_wmma_f32_16x16x32_bf16(false, a, false, bm,
                                                        (short)0, c, false, false);
        }
        __syncthreads();
    }
    // epilogue: C layout -> lanes 0-15: M=r, lanes 16-31: M=8+r; N = lane&15
    const int n     = n0 + (lane & 15);
    const int mbase = mt * 16 + ((lane < 16) ? 0 : 8);
    const float bias = b1[n];
    #pragma unroll
    for (int r = 0; r < 8; ++r) {
        float v = tanhf(c[r] + bias);
        feat[(size_t)(mbase + r) * D_ + n] = f2bf(v);
    }
}

// Partial softmax GEMM with online (max,sumexp) accumulation.
// M=32 per block: each B fragment feeds TWO WMMAs (32 FLOP per L2 byte).
// grid (L/32, 50), block 256 (8 waves). Each wave: 5 n-tiles x 32 K-steps x 2 = 320 WMMAs.
__global__ void __launch_bounds__(256, 1)
gemm2_partial_kernel(const __bf16* __restrict__ feat,
                     const __bf16* __restrict__ Wb,
                     float* __restrict__ partials) {
    __shared__ __bf16 As[32 * 1024];       // 64 KB: full K for 32 rows
    const int mt   = blockIdx.x;           // 0..127 (32 rows each)
    const int wave = threadIdx.x >> 5;
    const int lane = threadIdx.x & 31;

    const __bf16* srcA = feat + (size_t)mt * 32 * K2_;
#if ASYNC_LDS
    #pragma unroll 1
    for (int i = threadIdx.x; i < 32 * 64; i += 256) {
        int r = i >> 6, ch = i & 63;
        __builtin_amdgcn_global_load_async_to_lds_b128(
            GAS(srcA + (size_t)r * K2_ + ch * 16),
            LAS(As + r * 1024 + ch * 16), 0, 0);
    }
    __builtin_amdgcn_s_wait_asynccnt(0);
#else
    #pragma unroll 1
    for (int i = threadIdx.x; i < 32 * 64; i += 256) {
        ((v16bf*)As)[i] = *(const v16bf*)(srcA + (size_t)(i >> 6) * K2_ + (i & 63) * 16);
    }
#endif
    __syncthreads();

    const int row  = lane & 15;
    const int koff = (lane < 16) ? 0 : 16;
    const int vbase = blockIdx.y * 640 + wave * 80;

    // per-lane running softmax state: row-group 0 (M 0..15) and 1 (M 16..31)
    float rm0[8], rs0[8], rm1[8], rs1[8];
    #pragma unroll
    for (int r = 0; r < 8; ++r) {
        rm0[r] = -1e30f; rs0[r] = 0.0f;
        rm1[r] = -1e30f; rs1[r] = 0.0f;
    }

    #pragma unroll 1
    for (int t = 0; t < 5; ++t) {
        int n0 = vbase + t * 16;
        const __bf16* bp = Wb + (size_t)(n0 + row) * K2_ + koff;
        v8f c0 = {}, c1 = {};
        #pragma unroll 2
        for (int k = 0; k < K2_; k += 32) {
            v16bf bm = *(const v16bf*)(bp + k);
            v16bf a0 = *(const v16bf*)(As + row * 1024 + k + koff);
            v16bf a1 = *(const v16bf*)(As + (16 + row) * 1024 + k + koff);
            __builtin_prefetch((const void*)(bp + k + 256), 0, 1);
            c0 = __builtin_amdgcn_wmma_f32_16x16x32_bf16(false, a0, false, bm,
                                                         (short)0, c0, false, false);
            c1 = __builtin_amdgcn_wmma_f32_16x16x32_bf16(false, a1, false, bm,
                                                         (short)0, c1, false, false);
        }
        // online merge: each lane gets one column value per row r per group
        #pragma unroll
        for (int r = 0; r < 8; ++r) {
            float v = c0[r];
            float mn = fmaxf(rm0[r], v);
            rs0[r] = rs0[r] * __expf(rm0[r] - mn) + __expf(v - mn);
            rm0[r] = mn;
            float w = c1[r];
            float mw = fmaxf(rm1[r], w);
            rs1[r] = rs1[r] * __expf(rm1[r] - mw) + __expf(w - mw);
            rm1[r] = mw;
        }
    }

    // cross-lane combine of (m, s) pairs within each 16-lane half
    const int vtile = blockIdx.y * 8 + wave;       // 0..399
    const int half  = (lane < 16) ? 0 : 8;
    #pragma unroll
    for (int r = 0; r < 8; ++r) {
        float m0 = rm0[r], s0 = rs0[r];
        float m1 = rm1[r], s1 = rs1[r];
        #pragma unroll
        for (int x = 1; x < 16; x <<= 1) {
            float mo = __shfl_xor(m0, x, 32);
            float so = __shfl_xor(s0, x, 32);
            float mn = fmaxf(m0, mo);
            s0 = s0 * __expf(m0 - mn) + so * __expf(mo - mn);
            m0 = mn;
            mo = __shfl_xor(m1, x, 32);
            so = __shfl_xor(s1, x, 32);
            mn = fmaxf(m1, mo);
            s1 = s1 * __expf(m1 - mn) + so * __expf(mo - mn);
            m1 = mn;
        }
        if ((lane & 15) == 0) {
            size_t i0 = ((size_t)(mt * 32 + half + r) * NVT_ + vtile) * 2;
            partials[i0]     = m0;
            partials[i0 + 1] = s0;
            size_t i1 = ((size_t)(mt * 32 + 16 + half + r) * NVT_ + vtile) * 2;
            partials[i1]     = m1;
            partials[i1 + 1] = s1;
        }
    }
}

// tag_logit[l] = dot(feat_bf16[l], Wout_bf16[tags[l]])  (one wave per row)
__global__ void tag_logit_kernel(const __bf16* __restrict__ feat,
                                 const __bf16* __restrict__ Wb,
                                 const int* __restrict__ tags,
                                 float* __restrict__ tl) {
    int l = blockIdx.x * 8 + (threadIdx.x >> 5);
    int lane = threadIdx.x & 31;
    int tag = tags[l];
    const __bf16* f = feat + (size_t)l * D_;
    const __bf16* w = Wb + (size_t)tag * K2_;
    float s = 0.0f;
    #pragma unroll 1
    for (int ch = lane; ch < 64; ch += 32) {      // 64 chunks of 16 bf16
        v16bf fv = *(const v16bf*)(f + ch * 16);
        v16bf wv = *(const v16bf*)(w + ch * 16);
        #pragma unroll
        for (int j = 0; j < 16; ++j) s = fmaf(bf2f(fv[j]), bf2f(wv[j]), s);
    }
    for (int m = 1; m < 32; m <<= 1) s += __shfl_xor(s, m, 32);
    if (lane == 0) tl[l] = s;
}

// Combine partials -> logsumexp; nll = lse - tag_logit; weighted sum into out.
__global__ void finalize_kernel(const float* __restrict__ partials,
                                const float* __restrict__ tl,
                                const int* __restrict__ tags,
                                const float* __restrict__ dp,
                                float* __restrict__ out) {
    int l = blockIdx.x * 8 + (threadIdx.x >> 5);
    int lane = threadIdx.x & 31;
    const float* p = partials + (size_t)l * NVT_ * 2;
    float m = -1e30f;
    #pragma unroll 1
    for (int i = lane; i < NVT_; i += 32) m = fmaxf(m, p[2 * i]);
    for (int x = 1; x < 32; x <<= 1) m = fmaxf(m, __shfl_xor(m, x, 32));
    float s = 0.0f;
    #pragma unroll 1
    for (int i = lane; i < NVT_; i += 32) s += p[2 * i + 1] * __expf(p[2 * i] - m);
    for (int x = 1; x < 32; x <<= 1) s += __shfl_xor(s, x, 32);
    if (lane == 0) {
        float lse = m + logf(s);
        float nll = lse - tl[l];
        float scaled = 1.0f - dp[tags[l]];
        atomicAdd(out, nll * scaled * (1.0f / ((float)L_ + 1e-5f)));
    }
}

extern "C" void kernel_launch(void* const* d_in, const int* in_sizes, int n_in,
                              void* d_out, int out_size, void* d_ws, size_t ws_size,
                              hipStream_t stream) {
    (void)in_sizes; (void)n_in; (void)out_size; (void)ws_size;
    const float* hidden  = (const float*)d_in[0];   // [S,B,H]
    const float* W1      = (const float*)d_in[1];   // [D,2H]
    const float* b1      = (const float*)d_in[2];   // [D]
    const float* Wout    = (const float*)d_in[3];   // [V,D]
    const float* dp      = (const float*)d_in[4];   // [V]
    const int*   sbid    = (const int*)d_in[5];
    const int*   sbeg    = (const int*)d_in[6];
    const int*   sendm1  = (const int*)d_in[7];
    const int*   tags    = (const int*)d_in[8];
    float* out = (float*)d_out;

    char* ws = (char*)d_ws;
    // workspace layout (all offsets 256B-aligned)
    __bf16* Wout_bf  = (__bf16*)(ws);                               //  65,536,000 B
    __bf16* W1_bf    = (__bf16*)(ws + 65536000ull);                 //   4,194,304 B
    __bf16* span_bf  = (__bf16*)(ws + 69730304ull);                 //  16,777,216 B
    __bf16* feat_bf  = (__bf16*)(ws + 86507520ull);                 //   8,388,608 B
    float*  partials = (float*)(ws + 94896128ull);                  //  13,107,200 B
    float*  taglog   = (float*)(ws + 108003328ull);                 //      16,384 B

    zero_out_kernel<<<1, 1, 0, stream>>>(out);

    cvt_f32_bf16_kernel<<<2048, 256, 0, stream>>>(Wout, Wout_bf, (long)V_ * D_);
    cvt_f32_bf16_kernel<<<1024, 256, 0, stream>>>(W1, W1_bf, (long)D_ * K1_);

    gather_kernel<<<L_, 256, 0, stream>>>(hidden, sbid, sbeg, sendm1, span_bf);

    gemm1_kernel<<<dim3(L_ / 16, D_ / 128), 256, 0, stream>>>(span_bf, W1_bf, b1, feat_bf);

    gemm2_partial_kernel<<<dim3(L_ / 32, V_ / 640), 256, 0, stream>>>(feat_bf, Wout_bf, partials);

    tag_logit_kernel<<<L_ / 8, 256, 0, stream>>>(feat_bf, Wout_bf, tags, taglog);

    finalize_kernel<<<L_ / 8, 256, 0, stream>>>(partials, taglog, tags, dp, out);
}
